// STFT_46986942218955
// MI455X (gfx1250) — matrix operand order repaired and checked by reading, not compiled, not run on hardware
//
#include <hip/hip_runtime.h>
#include <hip/hip_bf16.h>
#include <stddef.h>

typedef __bf16 bf16x8  __attribute__((ext_vector_type(8)));
typedef __bf16 bf16x16 __attribute__((ext_vector_type(16)));
typedef float  f32x8   __attribute__((ext_vector_type(8)));
typedef unsigned int v4u __attribute__((ext_vector_type(4)));
typedef int v4i __attribute__((ext_vector_type(4)));
typedef int v8i __attribute__((ext_vector_type(8)));

#define NFFT   2048
#define HOP    512
#define KBINS  1025
#define NCOMB  2050              // real | imag fused along N
#define LSIG   320000
#define TFR    626
#define MTOT   (16 * TFR)        // 10016
#define RESZ   ((long)MTOT * KBINS)

#define XPLEN  (LSIG + NFFT)     // 322048 padded samples per batch
#define NPADW  2064              // Wt rows, zero-padded past 2050

// ---- main GEMM tiling ----
#define BM 64                    // frames per workgroup tile (within one batch)
#define BN 128                   // fused output columns per tile
#define KC 32                    // K chunk = one WMMA K
#define TTILES ((TFR + BM - 1) / BM)   // 10 t-tiles per batch
#define LSTR 40                  // LDS row stride in bf16 elems (32 data + 8 pad, via TDM pad)

// per-buffer LDS element offsets
#define OFF_AHI 0
#define OFF_ALO (BM * LSTR)              // 2560
#define OFF_BHI (2 * BM * LSTR)          // 5120
#define OFF_BLO (2 * BM * LSTR + BN * LSTR)  // 10240
#define BUFE    (2 * BM * LSTR + 2 * BN * LSTR) // 15360 elems per buffer

// ---- workspace layout (bf16 elems) ----
#define S_XP ((size_t)16 * XPLEN)        // one xp plane
#define S_WT ((size_t)NPADW * NFFT)      // one Wt plane
#define WS_NEEDED ((2 * S_XP + 2 * S_WT) * sizeof(__bf16))

__device__ __forceinline__ void split_bf16(float v, __bf16 &hi, __bf16 &lo) {
    hi = (__bf16)v;
    lo = (__bf16)(v - (float)hi);
}

__device__ __forceinline__ bf16x16 join16(bf16x8 a, bf16x8 b) {
    return __builtin_shufflevector(a, b, 0,1,2,3,4,5,6,7,8,9,10,11,12,13,14,15);
}

// ============================================================================
// Prep 1: reflect-pad x and split into bf16 hi/lo planes
// ============================================================================
__global__ __launch_bounds__(256)
void prep_x(const float* __restrict__ x, __bf16* __restrict__ xphi,
            __bf16* __restrict__ xplo)
{
    long i = (long)blockIdx.x * 256 + threadIdx.x;
    if (i >= (long)16 * XPLEN) return;
    int b = (int)(i / XPLEN);
    int p = (int)(i - (long)b * XPLEN);
    int j = p - (NFFT / 2);
    int src = (j < 0) ? -j : ((j >= LSIG) ? (2 * LSIG - 2 - j) : j);
    float v = x[(long)b * LSIG + src];
    __bf16 hi, lo; split_bf16(v, hi, lo);
    xphi[i] = hi;
    xplo[i] = lo;
}

// ============================================================================
// Prep 2: fuse real|imag, transpose to Wt[n][k], split hi/lo, zero-pad rows
// ============================================================================
__global__ __launch_bounds__(256)
void prep_w(const float* __restrict__ wr, const float* __restrict__ wi,
            __bf16* __restrict__ wthi, __bf16* __restrict__ wtlo)
{
    long i = (long)blockIdx.x * 256 + threadIdx.x;
    if (i >= (long)NPADW * NFFT) return;
    int n = (int)(i / NFFT);
    int k = (int)(i - (long)n * NFFT);
    float v = 0.f;
    if (n < KBINS)      v = wr[(long)k * KBINS + n];
    else if (n < NCOMB) v = wi[(long)k * KBINS + (n - KBINS)];
    __bf16 hi, lo; split_bf16(v, hi, lo);
    wthi[i] = hi;
    wtlo[i] = lo;
}

// ============================================================================
// Tensor Data Mover: 2D strided tile -> LDS, with LDS row padding 32+8 elems.
// D# bit layout per CDNA5 ISA 8.3-8.6.  data_size=2B, pad_interval=16 DWORDs,
// pad_amount=4 DWORDs  => LDS row stride = 80 B = LSTR bf16 elems.
// tensor_dim0 == tile_w (no X OOB); tensor_h < tile_h zero-fills edge rows.
// ============================================================================
__device__ __forceinline__ void tdm_load_2d(unsigned lds_off_bytes,
                                            const __bf16* gsrc,
                                            unsigned tile_w, unsigned tile_h,
                                            unsigned tensor_h,
                                            unsigned row_stride_elems)
{
    unsigned long long ga = (unsigned long long)(size_t)gsrc;
    v4u g0;
    g0.x = 1u;                                          // count=1, user mode
    g0.y = lds_off_bytes;                               // lds_addr
    g0.z = (unsigned)(ga & 0xffffffffu);                // global_addr[31:0]
    g0.w = (unsigned)((ga >> 32) & 0x1ffffffu) | (2u << 30);  // addr[56:32] | type=2
    v8i g1;
    g1[0] = (int)((1u << 16)        // data_size = 2 bytes
                | (1u << 20)        // pad_enable
                | (3u << 22)        // pad_interval: 16 DWORDs
                | (3u << 25));      // pad_amount:   4 DWORDs
    g1[1] = (int)((tile_w & 0xffffu) << 16);                            // tensor_dim0 lo
    g1[2] = (int)(((tile_w >> 16) & 0xffffu) | ((tensor_h & 0xffffu) << 16)); // dim0 hi | dim1 lo
    g1[3] = (int)(((tensor_h >> 16) & 0xffffu) | ((tile_w & 0xffffu) << 16)); // dim1 hi | tile_dim0
    g1[4] = (int)(tile_h & 0xffffu);                                    // tile_dim1 | tile_dim2=0
    g1[5] = (int)row_stride_elems;                                      // tensor_dim0_stride lo
    g1[6] = 0;                                                          // stride hi | dim1_stride lo
    g1[7] = 0;
    v4i z = {0, 0, 0, 0};
#if __clang_major__ >= 23
    v8i z8 = {0, 0, 0, 0, 0, 0, 0, 0};
    __builtin_amdgcn_tensor_load_to_lds(g0, g1, z, z, z8, 0);
#else
    __builtin_amdgcn_tensor_load_to_lds(g0, g1, z, z, 0);
#endif
}

// ============================================================================
// Main GEMM:  out[b,t,j] = sum_k frames(b,t,k) * Wt(j,k),  bf16x3 split,
// TDM double-buffered staging, 12 WMMAs / chunk / wave.
// ============================================================================
__global__ __launch_bounds__(256)
void stft_wmma_tdm(const __bf16* __restrict__ xphi,
                   const __bf16* __restrict__ xplo,
                   const __bf16* __restrict__ wthi,
                   const __bf16* __restrict__ wtlo,
                   float* __restrict__ out)
{
    __shared__ __bf16 lds[2 * BUFE];

    const int tid  = threadIdx.x;
    const int lane = tid & 31;
    const int wv   = tid >> 5;
    const int wm   = wv & 1;          // 2 wave-rows  -> M sub-range of 32
    const int wn   = wv >> 1;         // 4 wave-cols  -> N sub-range of 32
    const int h    = lane >> 4;
    const int ln   = lane & 15;

    const int n0 = blockIdx.x * BN;
    const int bt = blockIdx.y;
    const int b  = bt / TTILES;
    const int t0 = (bt % TTILES) * BM;
    const int validA = (TFR - t0 < BM) ? (TFR - t0) : BM;
    const int validB = (NPADW - n0 < BN) ? (NPADW - n0) : BN;

    const __bf16* aHs = xphi + (size_t)b * XPLEN + (size_t)t0 * HOP;
    const __bf16* aLs = xplo + (size_t)b * XPLEN + (size_t)t0 * HOP;
    const __bf16* bHs = wthi + (size_t)n0 * NFFT;
    const __bf16* bLs = wtlo + (size_t)n0 * NFFT;

    const unsigned ldsBase = (unsigned)(size_t)(void*)&lds[0];

    // issue chunk 0 into buffer 0 (wave 0 only; TDM ignores EXEC, one DMA per op)
    if (wv == 0) {
        unsigned base = ldsBase;
        tdm_load_2d(base + OFF_AHI * 2, aHs, KC, BM, (unsigned)validA, HOP);
        tdm_load_2d(base + OFF_ALO * 2, aLs, KC, BM, (unsigned)validA, HOP);
        tdm_load_2d(base + OFF_BHI * 2, bHs, KC, BN, (unsigned)validB, NFFT);
        tdm_load_2d(base + OFF_BLO * 2, bLs, KC, BN, (unsigned)validB, NFFT);
    }

    f32x8 acc[2][2] = {};

    const int NITER = NFFT / KC;      // 64
    for (int it = 0; it < NITER; ++it) {
        if (wv == 0) __builtin_amdgcn_s_wait_tensorcnt(0);
        __syncthreads();              // buffer (it&1) ready; prev compute drained

        if (wv == 0 && it + 1 < NITER) {
            int k1 = (it + 1) * KC;
            unsigned base = ldsBase + ((it + 1) & 1) * BUFE * 2;
            tdm_load_2d(base + OFF_AHI * 2, aHs + k1, KC, BM, (unsigned)validA, HOP);
            tdm_load_2d(base + OFF_ALO * 2, aLs + k1, KC, BM, (unsigned)validA, HOP);
            tdm_load_2d(base + OFF_BHI * 2, bHs + k1, KC, BN, (unsigned)validB, NFFT);
            tdm_load_2d(base + OFF_BLO * 2, bLs + k1, KC, BN, (unsigned)validB, NFFT);
        }

        const __bf16* buf = &lds[(it & 1) * BUFE];

        // A fragments: documented 16-bit A layout (K = h*8..+7 then 16+h*8..+7)
        bf16x16 aH[2], aL[2];
        #pragma unroll
        for (int sm = 0; sm < 2; ++sm) {
            const __bf16* ah = buf + OFF_AHI + (wm * 32 + sm * 16 + ln) * LSTR;
            const __bf16* al = buf + OFF_ALO + (wm * 32 + sm * 16 + ln) * LSTR;
            aH[sm] = join16(*(const bf16x8*)(ah + h * 8),
                            *(const bf16x8*)(ah + 16 + h * 8));
            aL[sm] = join16(*(const bf16x8*)(al + h * 8),
                            *(const bf16x8*)(al + 16 + h * 8));
        }
        // B fragments: per-lane column n, K = h*8..+7 then 16+h*8..+7 (matches A)
        bf16x16 bH[2], bL[2];
        #pragma unroll
        for (int sn = 0; sn < 2; ++sn) {
            const __bf16* bh = buf + OFF_BHI + (wn * 32 + sn * 16 + ln) * LSTR;
            const __bf16* bl = buf + OFF_BLO + (wn * 32 + sn * 16 + ln) * LSTR;
            bH[sn] = join16(*(const bf16x8*)(bh + h * 8),
                            *(const bf16x8*)(bh + 16 + h * 8));
            bL[sn] = join16(*(const bf16x8*)(bl + h * 8),
                            *(const bf16x8*)(bl + 16 + h * 8));
        }

        // 12 WMMAs: bf16x3 split GEMM (hi*hi + hi*lo + lo*hi), f32 accumulate
        #pragma unroll
        for (int sm = 0; sm < 2; ++sm)
            #pragma unroll
            for (int sn = 0; sn < 2; ++sn) {
                acc[sm][sn] = __builtin_amdgcn_wmma_f32_16x16x32_bf16(
                    false, aH[sm], false, bH[sn], (short)0, acc[sm][sn], false, false);
                acc[sm][sn] = __builtin_amdgcn_wmma_f32_16x16x32_bf16(
                    false, aH[sm], false, bL[sn], (short)0, acc[sm][sn], false, false);
                acc[sm][sn] = __builtin_amdgcn_wmma_f32_16x16x32_bf16(
                    false, aL[sm], false, bH[sn], (short)0, acc[sm][sn], false, false);
            }
        // no trailing barrier: next iteration's barrier orders reuse of this buffer
    }

    // epilogue: C/D layout  VGPR r -> M = r + 8*h,  N = lane&15
    #pragma unroll
    for (int sm = 0; sm < 2; ++sm) {
        #pragma unroll
        for (int sn = 0; sn < 2; ++sn) {
            const int j = n0 + wn * 32 + sn * 16 + ln;
            #pragma unroll
            for (int r = 0; r < 8; ++r) {
                int t = t0 + wm * 32 + sm * 16 + h * 8 + r;
                if (t >= TFR) continue;
                long gm = (long)b * TFR + t;
                float v = acc[sm][sn][r];
                if (j < KBINS)      out[gm * KBINS + j] = v;
                else if (j < NCOMB) out[RESZ + gm * KBINS + (j - KBINS)] = v;
            }
        }
    }
}

// ============================================================================
// Fallback (round-1 kernel): conversion in-kernel, no workspace required.
// ============================================================================
#define FBM 32
#define FBN 64
#define AST 40
#define WST 40

__global__ __launch_bounds__(256)
void stft_wmma_fallback(const float* __restrict__ x,
                        const float* __restrict__ wr,
                        const float* __restrict__ wi,
                        float* __restrict__ out)
{
    __shared__ __bf16 Ahi[FBM * AST];
    __shared__ __bf16 Alo[FBM * AST];
    __shared__ __bf16 Bhi[FBN * WST];
    __shared__ __bf16 Blo[FBN * WST];

    const int tid  = threadIdx.x;
    const int lane = tid & 31;
    const int wv   = tid >> 5;
    const int mi   = wv & 1;
    const int ni   = wv >> 1;
    const int h    = lane >> 4;
    const int ln   = lane & 15;

    const int m0 = blockIdx.y * FBM;
    const int n0 = blockIdx.x * FBN;

    const int arow  = tid >> 3;
    const int acol0 = (tid & 7) * 4;
    const int m_g   = m0 + arow;
    const int t_f   = m_g % TFR;
    const int b_f   = m_g / TFR;
    const long xbase = (long)b_f * LSIG;
    const int posb   = t_f * HOP - (NFFT / 2);

    const int wrow  = tid >> 3;
    const int wcol0 = (tid & 7) * 8;

    f32x8 acc = {};

    for (int k0 = 0; k0 < NFFT; k0 += KC) {
        if (k0 + KC < NFFT)
            __builtin_prefetch(&wr[(long)(k0 + KC + wrow) * KBINS + n0], 0, 1);

        #pragma unroll
        for (int c = 0; c < 4; ++c) {
            int j  = posb + k0 + acol0 + c;
            int src = (j < 0) ? -j : ((j >= LSIG) ? (2 * LSIG - 2 - j) : j);
            __bf16 hi, lo; split_bf16(x[xbase + src], hi, lo);
            Ahi[arow * AST + acol0 + c] = hi;
            Alo[arow * AST + acol0 + c] = lo;
        }
        const int krow = k0 + wrow;
        #pragma unroll
        for (int c = 0; c < 8; ++c) {
            int j = n0 + wcol0 + c;
            float v = 0.f;
            if (j < KBINS)      v = wr[(long)krow * KBINS + j];
            else if (j < NCOMB) v = wi[(long)krow * KBINS + (j - KBINS)];
            __bf16 hi, lo; split_bf16(v, hi, lo);
            Bhi[(wcol0 + c) * WST + wrow] = hi;
            Blo[(wcol0 + c) * WST + wrow] = lo;
        }
        __syncthreads();

        const __bf16* ah = &Ahi[(mi * 16 + ln) * AST];
        const __bf16* al = &Alo[(mi * 16 + ln) * AST];
        bf16x16 aH = join16(*(const bf16x8*)(ah + h * 8), *(const bf16x8*)(ah + 16 + h * 8));
        bf16x16 aL = join16(*(const bf16x8*)(al + h * 8), *(const bf16x8*)(al + 16 + h * 8));
        const __bf16* bh = &Bhi[(ni * 16 + ln) * WST];
        const __bf16* bl = &Blo[(ni * 16 + ln) * WST];
        bf16x16 bH = join16(*(const bf16x8*)(bh + h * 8), *(const bf16x8*)(bh + 16 + h * 8));
        bf16x16 bL = join16(*(const bf16x8*)(bl + h * 8), *(const bf16x8*)(bl + 16 + h * 8));

        acc = __builtin_amdgcn_wmma_f32_16x16x32_bf16(false, aH, false, bH, (short)0, acc, false, false);
        acc = __builtin_amdgcn_wmma_f32_16x16x32_bf16(false, aH, false, bL, (short)0, acc, false, false);
        acc = __builtin_amdgcn_wmma_f32_16x16x32_bf16(false, aL, false, bH, (short)0, acc, false, false);
        __syncthreads();
    }

    const int j = n0 + ni * 16 + ln;
    #pragma unroll
    for (int r = 0; r < 8; ++r) {
        int gm = m0 + mi * 16 + h * 8 + r;
        float v = acc[r];
        if (j < KBINS)      out[(long)gm * KBINS + j] = v;
        else if (j < NCOMB) out[RESZ + (long)gm * KBINS + (j - KBINS)] = v;
    }
}

// ============================================================================
extern "C" void kernel_launch(void* const* d_in, const int* in_sizes, int n_in,
                              void* d_out, int out_size, void* d_ws, size_t ws_size,
                              hipStream_t stream) {
    (void)in_sizes; (void)n_in; (void)out_size;
    const float* x  = (const float*)d_in[0];
    const float* wr = (const float*)d_in[1];
    const float* wi = (const float*)d_in[2];
    float* out = (float*)d_out;

    if (ws_size >= WS_NEEDED) {
        __bf16* ws   = (__bf16*)d_ws;
        __bf16* xphi = ws;
        __bf16* xplo = ws + S_XP;
        __bf16* wthi = ws + 2 * S_XP;
        __bf16* wtlo = ws + 2 * S_XP + S_WT;

        long nx = (long)16 * XPLEN;
        prep_x<<<dim3((unsigned)((nx + 255) / 256)), dim3(256), 0, stream>>>(x, xphi, xplo);
        long nw = (long)NPADW * NFFT;
        prep_w<<<dim3((unsigned)((nw + 255) / 256)), dim3(256), 0, stream>>>(wr, wi, wthi, wtlo);

        dim3 grid((NCOMB + BN - 1) / BN,        // 17 N-tiles
                  16 * TTILES);                  // 160 (batch, t-tile) pairs
        stft_wmma_tdm<<<grid, dim3(256), 0, stream>>>(xphi, xplo, wthi, wtlo, out);
    } else {
        dim3 grid((NCOMB + FBN - 1) / FBN, MTOT / FBM);
        stft_wmma_fallback<<<grid, dim3(256), 0, stream>>>(x, wr, wi, out);
    }
}
